// MoE_53197464928568
// MI455X (gfx1250) — compile-verified, matrix operand order checked by reading, not run on hardware
//
#include <hip/hip_runtime.h>

// Problem constants (from reference): B=8, N=4096 -> T=32768 tokens
#define T_TOK 32768
#define DD 128     // model dim
#define HH 256     // hidden dim
#define EE 16      // experts
// top-k K = 2 (hardcoded in gating math)

typedef __attribute__((ext_vector_type(16))) __bf16 v16bf;
typedef __attribute__((ext_vector_type(8)))  float  v8f;

struct __attribute__((aligned(16))) U128 { unsigned int a, b, c, d; };
union Frag16 { v16bf v; U128 q[2]; };

__device__ __forceinline__ unsigned short f2bf(float f) {
  unsigned int u = __float_as_uint(f);
  u += 0x7FFFu + ((u >> 16) & 1u);           // round-to-nearest-even
  return (unsigned short)(u >> 16);
}

// Load one 16x32 bf16 A/B fragment (8 VGPRs) from a row-major [rows][rowStride]
// bf16 array. Per ISA 7.12.2 (16-bit A 16x32): lanes 0-15 hold row=lane,
// K = {kbase..kbase+7, kbase+16..kbase+23}; lanes 16-31 hold K offset by +8.
__device__ __forceinline__ v16bf load_frag(const unsigned short* base, int rowStride,
                                           int kbase, int lane) {
  const int m   = lane & 15;
  const int off = (lane < 16) ? 0 : 8;
  const unsigned short* p = base + m * rowStride + kbase + off;
  Frag16 f;
  f.q[0] = *(const U128*)(p);
  f.q[1] = *(const U128*)(p + 16);
  return f.v;
}

__device__ __forceinline__ v8f wmma_bf16(v16bf a, v16bf b, v8f c) {
  return __builtin_amdgcn_wmma_f32_16x16x32_bf16(false, a, false, b, (short)0, c,
                                                 false, false);
}

// ---- weight convert + transpose: src[b][r][c] (f32) -> dst[b][c][r] (bf16) ----
__global__ void convT_f32_to_bf16(const float* __restrict__ src,
                                  unsigned short* __restrict__ dst,
                                  int Bn, int R, int C) {
  long total = (long)Bn * R * C;
  long rc = (long)R * C;
  for (long i = (long)blockIdx.x * blockDim.x + threadIdx.x; i < total;
       i += (long)gridDim.x * blockDim.x) {
    int b = (int)(i / rc);
    int rem = (int)(i - (long)b * rc);
    int r = rem / C, c = rem - r * C;
    dst[(long)b * rc + (long)c * R + r] = f2bf(src[i]);
  }
}

__global__ void zero_cnt_kernel(int* cnt) {
  if (threadIdx.x < EE) cnt[threadIdx.x] = 0;
}

// ---- gating + routing + x->bf16 conversion ----
__global__ void __launch_bounds__(256) gate_route_kernel(
    const float* __restrict__ x, const float* __restrict__ Wg,
    const float* __restrict__ bg, unsigned short* __restrict__ xb,
    float* __restrict__ gates, float* __restrict__ omega,
    int* __restrict__ cnt, int* __restrict__ bucket) {
  __shared__ float wg_s[DD * EE];
  __shared__ float bg_s[EE];
  const int tid = threadIdx.x;
  for (int i = tid; i < DD * EE; i += 256) wg_s[i] = Wg[i];
  if (tid < EE) bg_s[tid] = bg[tid];
  __syncthreads();

  const int t = blockIdx.x * 256 + tid;
  float logits[EE];
#pragma unroll
  for (int e = 0; e < EE; ++e) logits[e] = bg_s[e];

  const float4* xp = (const float4*)(x + (size_t)t * DD);
  unsigned int* xbp = (unsigned int*)(xb + (size_t)t * DD);
  for (int d4 = 0; d4 < DD / 4; ++d4) {
    float4 v = xp[d4];
    xbp[d4 * 2 + 0] = (unsigned int)f2bf(v.x) | ((unsigned int)f2bf(v.y) << 16);
    xbp[d4 * 2 + 1] = (unsigned int)f2bf(v.z) | ((unsigned int)f2bf(v.w) << 16);
    const float* wr = &wg_s[d4 * 4 * EE];
#pragma unroll
    for (int e = 0; e < EE; ++e)
      logits[e] += v.x * wr[e] + v.y * wr[EE + e] + v.z * wr[2 * EE + e] +
                   v.w * wr[3 * EE + e];
  }
  // top-2 (first index wins ties, matching lax.top_k)
  float v0 = -3.0e38f, v1 = -3.0e38f; int i0 = 0, i1 = 0;
#pragma unroll
  for (int e = 0; e < EE; ++e) {
    float l = logits[e];
    if (l > v0)      { v1 = v0; i1 = i0; v0 = l; i0 = e; }
    else if (l > v1) { v1 = l; i1 = e; }
  }
  // softmax over {v0, v1} (others are -inf -> 0)
  float ex  = __expf(v1 - v0);
  float inv = 1.0f / (1.0f + ex);
  float g0 = inv, g1 = ex * inv;
  gates[t * 2 + 0] = g0;
  gates[t * 2 + 1] = g1;
  omega[t] = 1.0f - g0;               // g0 is the max gate
  int p0 = atomicAdd(&cnt[i0], 1);
  bucket[i0 * T_TOK + p0] = t * 2;
  int p1 = atomicAdd(&cnt[i1], 1);
  bucket[i1 * T_TOK + p1] = t * 2 + 1;
}

// ---- routed expert FFN: 16 gathered tokens per block, 8 waves, bf16 WMMA ----
__global__ void __launch_bounds__(256) expert_ffn_kernel(
    const unsigned short* __restrict__ xb, const unsigned short* __restrict__ W1b,
    const unsigned short* __restrict__ W2b, const float* __restrict__ b1,
    const float* __restrict__ b2, const int* __restrict__ cnt,
    const int* __restrict__ bucket, const float* __restrict__ gates,
    float* __restrict__ contrib) {
  const int e = blockIdx.x;
  const int nTok = cnt[e];
  const int t0 = blockIdx.y * 16;
  if (t0 >= nTok) return;
  const int valid = min(16, nTok - t0);

  __shared__ unsigned short xs[16 * DD];
  __shared__ unsigned short hs[16 * HH];
  __shared__ int toks[16];

  const int tid = threadIdx.x;
  if (tid < 16) toks[tid] = bucket[e * T_TOK + min(t0 + tid, nTok - 1)];
  __syncthreads();
  for (int i = tid; i < 16 * DD / 2; i += 256) {   // gather 16 bf16 x-rows
    int m = i >> 6;                                // DD/2 = 64 dwords per row
    int k2 = i & 63;
    ((unsigned int*)xs)[i] =
        ((const unsigned int*)(xb + (size_t)(toks[m] >> 1) * DD))[k2];
  }
  __syncthreads();

  const int wave = tid >> 5;
  const int lane = tid & 31;
  const int mhi  = (lane >> 4) << 3;   // 0 or 8: C/D row group per ISA layout
  const int ncl  = lane & 15;

  // layer 1: [16x128] @ [128x256] ; 2 N-tiles per wave, 4 K-steps
  const unsigned short* w1e = W1b + (size_t)e * HH * DD;  // [HH rows][DD] N-major
  v8f acc0 = {0.f, 0.f, 0.f, 0.f, 0.f, 0.f, 0.f, 0.f};
  v8f acc1 = acc0;
#pragma unroll
  for (int ks = 0; ks < DD; ks += 32) {
    v16bf a  = load_frag(xs, DD, ks, lane);
    v16bf bA = load_frag(w1e + (size_t)(wave * 2 + 0) * 16 * DD, DD, ks, lane);
    v16bf bB = load_frag(w1e + (size_t)(wave * 2 + 1) * 16 * DD, DD, ks, lane);
    acc0 = wmma_bf16(a, bA, acc0);
    acc1 = wmma_bf16(a, bB, acc1);
  }
  {
    int nc0 = (wave * 2 + 0) * 16 + ncl;
    int nc1 = (wave * 2 + 1) * 16 + ncl;
    float bb0 = b1[e * HH + nc0];
    float bb1 = b1[e * HH + nc1];
#pragma unroll
    for (int r = 0; r < 8; ++r) {
      float u0 = acc0[r] + bb0; u0 = u0 > 0.f ? u0 : 0.f;
      float u1 = acc1[r] + bb1; u1 = u1 > 0.f ? u1 : 0.f;
      hs[(mhi + r) * HH + nc0] = f2bf(u0);
      hs[(mhi + r) * HH + nc1] = f2bf(u1);
    }
  }
  __syncthreads();

  // layer 2: [16x256] @ [256x128] ; 1 N-tile per wave, 8 K-steps
  const unsigned short* w2e = W2b + (size_t)e * DD * HH;  // [DD rows][HH] N-major
  v8f acc2 = {0.f, 0.f, 0.f, 0.f, 0.f, 0.f, 0.f, 0.f};
#pragma unroll
  for (int ks = 0; ks < HH; ks += 32) {
    v16bf a = load_frag(hs, HH, ks, lane);
    v16bf b = load_frag(w2e + (size_t)wave * 16 * HH, HH, ks, lane);
    acc2 = wmma_bf16(a, b, acc2);
  }
  const int nc = wave * 16 + ncl;
  const float bias = b2[e * DD + nc];
#pragma unroll
  for (int r = 0; r < 8; ++r) {
    int m = mhi + r;
    if (m < valid) {
      int tk = toks[m];
      int token = tk >> 1, slot = tk & 1;
      float g = gates[token * 2 + slot];
      contrib[(size_t)slot * T_TOK * DD + (size_t)token * DD + nc] =
          g * (acc2[r] + bias);
    }
  }
}

// ---- universal expert over all tokens; writes omega*uni directly to out ----
__global__ void __launch_bounds__(256) universal_ffn_kernel(
    const unsigned short* __restrict__ xb, const unsigned short* __restrict__ Wu1b,
    const unsigned short* __restrict__ Wu2b, const float* __restrict__ bu1,
    const float* __restrict__ bu2, const float* __restrict__ omega,
    float* __restrict__ out) {
  const int t0 = blockIdx.x * 16;
  __shared__ unsigned short xs[16 * DD];
  __shared__ unsigned short hs[16 * HH];
  const int tid = threadIdx.x;
  const unsigned int* src = (const unsigned int*)(xb + (size_t)t0 * DD);
  for (int i = tid; i < 16 * DD / 2; i += 256) ((unsigned int*)xs)[i] = src[i];
  __syncthreads();

  const int wave = tid >> 5;
  const int lane = tid & 31;
  const int mhi  = (lane >> 4) << 3;
  const int ncl  = lane & 15;

  v8f acc0 = {0.f, 0.f, 0.f, 0.f, 0.f, 0.f, 0.f, 0.f};
  v8f acc1 = acc0;
#pragma unroll
  for (int ks = 0; ks < DD; ks += 32) {
    v16bf a  = load_frag(xs, DD, ks, lane);
    v16bf bA = load_frag(Wu1b + (size_t)(wave * 2 + 0) * 16 * DD, DD, ks, lane);
    v16bf bB = load_frag(Wu1b + (size_t)(wave * 2 + 1) * 16 * DD, DD, ks, lane);
    acc0 = wmma_bf16(a, bA, acc0);
    acc1 = wmma_bf16(a, bB, acc1);
  }
  {
    int nc0 = (wave * 2 + 0) * 16 + ncl;
    int nc1 = (wave * 2 + 1) * 16 + ncl;
    float bb0 = bu1[nc0], bb1 = bu1[nc1];
#pragma unroll
    for (int r = 0; r < 8; ++r) {
      float u0 = acc0[r] + bb0; u0 = u0 > 0.f ? u0 : 0.f;
      float u1 = acc1[r] + bb1; u1 = u1 > 0.f ? u1 : 0.f;
      hs[(mhi + r) * HH + nc0] = f2bf(u0);
      hs[(mhi + r) * HH + nc1] = f2bf(u1);
    }
  }
  __syncthreads();

  v8f acc2 = {0.f, 0.f, 0.f, 0.f, 0.f, 0.f, 0.f, 0.f};
#pragma unroll
  for (int ks = 0; ks < HH; ks += 32) {
    v16bf a = load_frag(hs, HH, ks, lane);
    v16bf b = load_frag(Wu2b + (size_t)wave * 16 * HH, HH, ks, lane);
    acc2 = wmma_bf16(a, b, acc2);
  }
  const int nc = wave * 16 + ncl;
  const float bias = bu2[nc];
#pragma unroll
  for (int r = 0; r < 8; ++r) {
    int m = mhi + r;
    out[(size_t)(t0 + m) * DD + nc] = omega[t0 + m] * (acc2[r] + bias);
  }
}

// ---- final combine: out += contrib_slot0 + contrib_slot1 ----
__global__ void combine_kernel(const float* __restrict__ contrib,
                               float* __restrict__ out) {
  const size_t total = (size_t)T_TOK * DD;
  for (size_t i = (size_t)blockIdx.x * blockDim.x + threadIdx.x; i < total;
       i += (size_t)gridDim.x * blockDim.x)
    out[i] += contrib[i] + contrib[total + i];
}

extern "C" void kernel_launch(void* const* d_in, const int* in_sizes, int n_in,
                              void* d_out, int out_size, void* d_ws, size_t ws_size,
                              hipStream_t stream) {
  (void)in_sizes; (void)n_in; (void)out_size; (void)ws_size;
  const float* x   = (const float*)d_in[0];
  const float* Wg  = (const float*)d_in[1];
  const float* bg  = (const float*)d_in[2];
  const float* W1  = (const float*)d_in[3];
  const float* b1  = (const float*)d_in[4];
  const float* W2  = (const float*)d_in[5];
  const float* b2  = (const float*)d_in[6];
  const float* Wu1 = (const float*)d_in[7];
  const float* bu1 = (const float*)d_in[8];
  const float* Wu2 = (const float*)d_in[9];
  const float* bu2 = (const float*)d_in[10];
  float* out = (float*)d_out;

  // workspace carve-up (~46.6 MB total), 256B aligned slices
  char* ws = (char*)d_ws;
  size_t off = 0;
  auto alloc = [&](size_t bytes) -> void* {
    void* p = ws + off;
    off += (bytes + 255) & ~(size_t)255;
    return p;
  };
  unsigned short* xb     = (unsigned short*)alloc((size_t)T_TOK * DD * 2);
  unsigned short* W1b    = (unsigned short*)alloc((size_t)EE * DD * HH * 2);
  unsigned short* W2b    = (unsigned short*)alloc((size_t)EE * DD * HH * 2);
  unsigned short* Wu1b   = (unsigned short*)alloc((size_t)DD * HH * 2);
  unsigned short* Wu2b   = (unsigned short*)alloc((size_t)DD * HH * 2);
  float*          gates  = (float*)alloc((size_t)T_TOK * 2 * 4);
  float*          omega  = (float*)alloc((size_t)T_TOK * 4);
  int*            cnt    = (int*)alloc(256);
  int*            bucket = (int*)alloc((size_t)EE * T_TOK * 4);
  float*          contrib= (float*)alloc((size_t)2 * T_TOK * DD * 4);

  const int totE = EE * DD * HH, totU = DD * HH;
  convT_f32_to_bf16<<<(totE + 255) / 256, 256, 0, stream>>>(W1, W1b, EE, DD, HH);
  convT_f32_to_bf16<<<(totE + 255) / 256, 256, 0, stream>>>(W2, W2b, EE, HH, DD);
  convT_f32_to_bf16<<<(totU + 255) / 256, 256, 0, stream>>>(Wu1, Wu1b, 1, DD, HH);
  convT_f32_to_bf16<<<(totU + 255) / 256, 256, 0, stream>>>(Wu2, Wu2b, 1, HH, DD);

  zero_cnt_kernel<<<1, 32, 0, stream>>>(cnt);
  gate_route_kernel<<<T_TOK / 256, 256, 0, stream>>>(x, Wg, bg, xb, gates, omega,
                                                     cnt, bucket);
  expert_ffn_kernel<<<dim3(EE, T_TOK / 16), 256, 0, stream>>>(
      xb, W1b, W2b, b1, b2, cnt, bucket, gates, contrib);
  universal_ffn_kernel<<<T_TOK / 16, 256, 0, stream>>>(xb, Wu1b, Wu2b, bu1, bu2,
                                                       omega, out);
  combine_kernel<<<2048, 256, 0, stream>>>(contrib, out);
}
